// SharedMolecularEncoder_31851477467898
// MI455X (gfx1250) — compile-verified, matrix operand order checked by reading, not run on hardware
//
#include <hip/hip_runtime.h>
#include <math.h>

// ---------------- problem constants (match reference) ----------------
#define N_ATOMS   100000
#define N_EDGES   200000
#define N_MOLS    5000
#define ATOM_DIM  39
#define BOND_DIM  10
#define HDIM      256
#define EDGE_DIM  64        // H/4
#define NODE_DIM  64
#define G3        (3*HDIM)  // 768
#define ATOM_PAD  64        // ATOM_DIM padded to k-step multiple
#define BOND_PAD  32        // BOND_DIM padded to k-step multiple

// ---------------- WMMA types ----------------
typedef __bf16 bf16x16 __attribute__((ext_vector_type(16)));
typedef float  f32x8   __attribute__((ext_vector_type(8)));

// ordered-uint encoding of float for atomicMax-based segment max
__device__ __forceinline__ unsigned fenc(float f) {
    union { float f; unsigned u; } v; v.f = f;
    return (v.u & 0x80000000u) ? ~v.u : (v.u | 0x80000000u);
}
__device__ __forceinline__ float fdec(unsigned u) {
    union { float f; unsigned x; } v;
    v.x = (u & 0x80000000u) ? (u & 0x7fffffffu) : ~u;
    return v.f;
}
#define ENC_NEGINF 0x007FFFFFu

// ---------------------------------------------------------------------
// Branchless fragment load: p = row_base + k0 + khalf (16B aligned, whole
// 32-wide k-window in bounds). 4x b128 + 8x v_cvt_pk_bf16_f32.
// ISA layout: f[0..7] = p[0..7], f[8..15] = p[16..23].
// ---------------------------------------------------------------------
__device__ __forceinline__ bf16x16 ldfrag(const float* __restrict__ p)
{
    const float4* q0 = (const float4*)p;
    const float4* q1 = (const float4*)(p + 16);
    float4 x0 = q0[0], x1 = q0[1];
    float4 x2 = q1[0], x3 = q1[1];
    bf16x16 f;
    f[0]  = (__bf16)x0.x; f[1]  = (__bf16)x0.y; f[2]  = (__bf16)x0.z; f[3]  = (__bf16)x0.w;
    f[4]  = (__bf16)x1.x; f[5]  = (__bf16)x1.y; f[6]  = (__bf16)x1.z; f[7]  = (__bf16)x1.w;
    f[8]  = (__bf16)x2.x; f[9]  = (__bf16)x2.y; f[10] = (__bf16)x2.z; f[11] = (__bf16)x2.w;
    f[12] = (__bf16)x3.x; f[13] = (__bf16)x3.y; f[14] = (__bf16)x3.z; f[15] = (__bf16)x3.w;
    return f;
}

// =====================================================================
// WMMA GEMM:  C[M,Nc] = act( A[M,K] @ W[Nc,K]^T + bias )
// Preconditions (guaranteed by caller): lda%4==0, (lda2%4==0 if A2),
// K a multiple of 32 expressed as ks1/ks2 step counts, Nc%32==0.
// - optional per-row gather (rowidx) on A (rows clamped for loads,
//   masked at store, so the inner loop is branch-free)
// - optional second K-segment A2 (GATEConv concat) as a second uniform
//   loop of ks2 steps.
// Wave tile: 64x32 C (4 M-subtiles x 2 N-subtiles); per k-step:
// 2 B-frags + 4 A-frags -> 8x v_wmma_f32_16x16x32_bf16.
// ACT: 0=none 1=leaky_relu(0.01) 2=relu
// =====================================================================
template <int ACT>
__global__ __launch_bounds__(256)
void gemm_fast_kernel(const float* __restrict__ A, int lda,
                      const int* __restrict__ rowidx,
                      const float* __restrict__ A2, int lda2,
                      int ks1, int ks2,
                      const float* __restrict__ Wt,   // [Nc, K] row major
                      const float* __restrict__ bias, // [Nc] or null
                      float* __restrict__ C,          // [M, Nc]
                      int M, int Nc, int K)
{
    const int lane = threadIdx.x & 31;
    const int wave = threadIdx.x >> 5;
    const int ntn  = Nc >> 5;                       // 32-col tiles
    const int ntm  = (M + 63) >> 6;                 // 64-row tiles
    const long tile = (long)blockIdx.x * 8 + wave;
    if (tile >= (long)ntm * ntn) return;            // wave-uniform
    const int mt = (int)(tile / ntn), nt = (int)(tile % ntn);
    const int m0 = mt << 6, n0 = nt << 5;

    const int lr    = lane & 15;
    const int khalf = (lane >> 4) << 3;             // 0 or 8

    const float* brow0 = Wt + (long)(n0 + lr) * K + khalf;
    const float* brow1 = Wt + (long)(n0 + 16 + lr) * K + khalf;

    const float* arp[4];
    const float* arp2[4];
#pragma unroll
    for (int r = 0; r < 4; ++r) {
        int ar = m0 + r * 16 + lr;
        if (ar >= M) ar = M - 1;                    // clamp: load valid, mask store
        const int ag = rowidx ? rowidx[ar] : ar;
        arp[r]  = A + (long)ag * lda + khalf;
        arp2[r] = A2 ? (A2 + (long)ar * lda2 + khalf) : (A + khalf);
    }

    f32x8 acc[4][2] = {};

    for (int ks = 0; ks < ks1; ++ks) {
        const int k0 = ks << 5;
        const bf16x16 b0 = ldfrag(brow0 + k0);
        const bf16x16 b1 = ldfrag(brow1 + k0);
#pragma unroll
        for (int r = 0; r < 4; ++r) {
            const bf16x16 a = ldfrag(arp[r] + k0);
            acc[r][0] = __builtin_amdgcn_wmma_f32_16x16x32_bf16(
                            false, a, false, b0, (short)0, acc[r][0], false, false);
            acc[r][1] = __builtin_amdgcn_wmma_f32_16x16x32_bf16(
                            false, a, false, b1, (short)0, acc[r][1], false, false);
        }
    }
    for (int ks = 0; ks < ks2; ++ks) {
        const int kb = (ks1 + ks) << 5;             // B offset continues in K
        const int ka = ks << 5;                     // A2 starts at 0
        const bf16x16 b0 = ldfrag(brow0 + kb);
        const bf16x16 b1 = ldfrag(brow1 + kb);
#pragma unroll
        for (int r = 0; r < 4; ++r) {
            const bf16x16 a = ldfrag(arp2[r] + ka);
            acc[r][0] = __builtin_amdgcn_wmma_f32_16x16x32_bf16(
                            false, a, false, b0, (short)0, acc[r][0], false, false);
            acc[r][1] = __builtin_amdgcn_wmma_f32_16x16x32_bf16(
                            false, a, false, b1, (short)0, acc[r][1], false, false);
        }
    }

    const int col0 = n0 + lr, col1 = n0 + 16 + lr;
    const float bv0 = bias ? bias[col0] : 0.f;
    const float bv1 = bias ? bias[col1] : 0.f;
#pragma unroll
    for (int r = 0; r < 4; ++r) {
#pragma unroll
        for (int q = 0; q < 8; ++q) {
            const int row = m0 + r * 16 + ((lane < 16) ? q : (8 + q));
            if (row < M) {
                float v0 = acc[r][0][q] + bv0;
                float v1 = acc[r][1][q] + bv1;
                if (ACT == 1)      { v0 = (v0 > 0.f) ? v0 : 0.01f * v0;
                                     v1 = (v1 > 0.f) ? v1 : 0.01f * v1; }
                else if (ACT == 2) { v0 = fmaxf(v0, 0.f); v1 = fmaxf(v1, 0.f); }
                C[(long)row * Nc + col0] = v0;
                C[(long)row * Nc + col1] = v1;
            }
        }
    }
}

// ---------------- row padding: out[r, 0:ldin] = in[r, :], rest 0 ----------------
__global__ void pad_rows_kernel(const float* __restrict__ in, int ldin,
                                float* __restrict__ out, int ldout, long total)
{
    long i = (long)blockIdx.x * blockDim.x + threadIdx.x;
    if (i >= total) return;
    int c = (int)(i % ldout);
    long r = i / ldout;
    out[i] = (c < ldin) ? in[r * ldin + c] : 0.f;
}

// ---------------- fill ----------------
__global__ void fill_f32_kernel(float* p, float v, long n) {
    long i = (long)blockIdx.x * blockDim.x + threadIdx.x;
    if (i < n) p[i] = v;
}
__global__ void fill_u32_kernel(unsigned* p, unsigned v, long n) {
    long i = (long)blockIdx.x * blockDim.x + threadIdx.x;
    if (i < n) p[i] = v;
}

// ---------------- wave-per-row dot: out[m] = dot(X[m,:D], v) ----------------
__global__ __launch_bounds__(256)
void rowdot_kernel(const float* __restrict__ X, const float* __restrict__ vv,
                   float* __restrict__ out, int M, int D)
{
    int row = blockIdx.x * 8 + (threadIdx.x >> 5);
    if (row >= M) return;
    int lane = threadIdx.x & 31;
    float s = 0.f;
    for (int j = lane; j < D; j += 32) s += X[(long)row * D + j] * vv[j];
    for (int o = 16; o; o >>= 1) s += __shfl_xor(s, o, 32);
    if (lane == 0) out[row] = s;
}

// GATEConv edge score: lrelu( dot(g[e], att_l) + dR[dst[e]] )
__global__ __launch_bounds__(256)
void gate_score_kernel(const float* __restrict__ g, const float* __restrict__ attl,
                       const float* __restrict__ dR, const int* __restrict__ dst,
                       float* __restrict__ out, int E, int D)
{
    int e = blockIdx.x * 8 + (threadIdx.x >> 5);
    if (e >= E) return;
    int lane = threadIdx.x & 31;
    float s = 0.f;
    for (int j = lane; j < D; j += 32) s += g[(long)e * D + j] * attl[j];
    for (int o = 16; o; o >>= 1) s += __shfl_xor(s, o, 32);
    if (lane == 0) {
        float v = s + dR[dst[e]];
        out[e] = (v > 0.f) ? v : 0.01f * v;
    }
}

// score[i] = lrelu( (idxA? a[idxA[i]] : a[i]) + b[idxB[i]] )
__global__ void score_kernel(const float* __restrict__ a, const int* __restrict__ idxA,
                             const float* __restrict__ b, const int* __restrict__ idxB,
                             float* __restrict__ out, int n)
{
    int i = blockIdx.x * blockDim.x + threadIdx.x;
    if (i >= n) return;
    float v = (idxA ? a[idxA[i]] : a[i]) + b[idxB[i]];
    out[i] = (v > 0.f) ? v : 0.01f * v;
}

// segment max via ordered-uint atomicMax
__global__ void segmax_kernel(const float* __restrict__ sc, const int* __restrict__ seg,
                              unsigned* __restrict__ mEnc, int n)
{
    int i = blockIdx.x * blockDim.x + threadIdx.x;
    if (i < n) atomicMax(&mEnc[seg[i]], fenc(sc[i]));
}

// w[i] = exp(sc - m[seg]); ssum[seg] += w
__global__ void segexp_kernel(const float* __restrict__ sc, const int* __restrict__ seg,
                              const unsigned* __restrict__ mEnc,
                              float* __restrict__ w, float* __restrict__ ssum, int n)
{
    int i = blockIdx.x * blockDim.x + threadIdx.x;
    if (i >= n) return;
    float e = __expf(sc[i] - fdec(mEnc[seg[i]]));
    w[i] = e;
    atomicAdd(&ssum[seg[i]], e);
}

// out[seg[i],:] += V[gather(i),:] * (w? w[i]/ssum[seg[i]] : 1)
__global__ __launch_bounds__(256)
void scatter_rows_kernel(const float* __restrict__ V, const int* __restrict__ gsrc,
                         const int* __restrict__ seg, const float* __restrict__ w,
                         const float* __restrict__ ssum, float* __restrict__ out,
                         int n, int D)
{
    int i = blockIdx.x * 8 + (threadIdx.x >> 5);
    if (i >= n) return;
    int lane = threadIdx.x & 31;
    int s = gsrc ? gsrc[i] : i;
    int d = seg[i];
    float a = w ? (w[i] / ssum[d]) : 1.0f;
    for (int j = lane; j < D; j += 32)
        atomicAdd(&out[(long)d * D + j], V[(long)s * D + j] * a);
}

// X[row,:] = act(X[row,:] + bias);  row = blockIdx.x, D = blockDim.x
// MODE 0 = elu, 1 = relu
template <int MODE>
__global__ void bias_act_kernel(float* __restrict__ X, const float* __restrict__ bias)
{
    long i = (long)blockIdx.x * blockDim.x + threadIdx.x;
    float v = X[i] + bias[threadIdx.x];
    X[i] = MODE ? fmaxf(v, 0.f) : ((v > 0.f) ? v : (__expf(v) - 1.f));
}

__global__ void relu_kernel(float* __restrict__ X, long n)
{
    long i = (long)blockIdx.x * blockDim.x + threadIdx.x;
    if (i < n) X[i] = fmaxf(X[i], 0.f);
}

// row = blockIdx.x, j = threadIdx.x (H==256)
// out = relu( (1-z)*tanh(i_n + r*h_n) + z*hprev )
__global__ void gru_combine_kernel(const float* __restrict__ gi, const float* __restrict__ gh,
                                   const float* __restrict__ hprev, float* __restrict__ out,
                                   int H)
{
    const int row = blockIdx.x, j = threadIdx.x;
    const float* gir = gi + (long)row * 3 * H;
    const float* ghr = gh + (long)row * 3 * H;
    float r = 1.f / (1.f + __expf(-(gir[j]         + ghr[j])));
    float z = 1.f / (1.f + __expf(-(gir[H + j]     + ghr[H + j])));
    float n = tanhf(gir[2 * H + j] + r * ghr[2 * H + j]);
    float h = hprev[(long)row * H + j];
    out[(long)row * H + j] = fmaxf((1.f - z) * n + z * h, 0.f);
}

// in-place LayerNorm over D=64, wave per row (2 elems/lane)
__global__ __launch_bounds__(256)
void layernorm_kernel(float* __restrict__ X, const float* __restrict__ g,
                      const float* __restrict__ b, int M)
{
    int row = blockIdx.x * 8 + (threadIdx.x >> 5);
    if (row >= M) return;
    int lane = threadIdx.x & 31;
    float a0 = X[(long)row * 64 + lane];
    float a1 = X[(long)row * 64 + 32 + lane];
    float s = a0 + a1;
    for (int o = 16; o; o >>= 1) s += __shfl_xor(s, o, 32);
    float mu = s * (1.f / 64.f);
    float d0 = a0 - mu, d1 = a1 - mu;
    float q = d0 * d0 + d1 * d1;
    for (int o = 16; o; o >>= 1) q += __shfl_xor(q, o, 32);
    float inv = rsqrtf(q * (1.f / 64.f) + 1e-5f);
    X[(long)row * 64 + lane]      = d0 * inv * g[lane]      + b[lane];
    X[(long)row * 64 + 32 + lane] = d1 * inv * g[32 + lane] + b[32 + lane];
}

// ---------------- host helpers ----------------
static inline long cdivl(long a, long b) { return (a + b - 1) / b; }

// fast GEMM: requires lda%4==0, (lda2%4==0), K%32==0, Nc%32==0
static inline void launch_fast(int act, const float* A, int lda, const int* ridx,
                               const float* A2, int lda2, int ksplit,
                               const float* Wt, const float* bias, float* C,
                               int M, int Nc, int K, hipStream_t s)
{
    const int ks1 = (A2 ? ksplit : K) / 32;
    const int ks2 = (A2 ? (K - ksplit) : 0) / 32;
    long tiles = cdivl(M, 64) * (Nc / 32);
    dim3 gr((unsigned)cdivl(tiles, 8)), bl(256);
    if (act == 0)
        gemm_fast_kernel<0><<<gr, bl, 0, s>>>(A, lda, ridx, A2, lda2, ks1, ks2, Wt, bias, C, M, Nc, K);
    else if (act == 1)
        gemm_fast_kernel<1><<<gr, bl, 0, s>>>(A, lda, ridx, A2, lda2, ks1, ks2, Wt, bias, C, M, Nc, K);
    else
        gemm_fast_kernel<2><<<gr, bl, 0, s>>>(A, lda, ridx, A2, lda2, ks1, ks2, Wt, bias, C, M, Nc, K);
}

static inline void fillf(float* p, float v, long n, hipStream_t s) {
    fill_f32_kernel<<<(unsigned)cdivl(n, 256), 256, 0, s>>>(p, v, n);
}
static inline void fillu(unsigned* p, unsigned v, long n, hipStream_t s) {
    fill_u32_kernel<<<(unsigned)cdivl(n, 256), 256, 0, s>>>(p, v, n);
}
static inline void padrows(const float* in, int ldin, float* out, int ldout,
                           long rows, hipStream_t s) {
    long total = rows * ldout;
    pad_rows_kernel<<<(unsigned)cdivl(total, 256), 256, 0, s>>>(in, ldin, out, ldout, total);
}

extern "C" void kernel_launch(void* const* d_in, const int* in_sizes, int n_in,
                              void* d_out, int out_size, void* d_ws, size_t ws_size,
                              hipStream_t stream)
{
    const int N = N_ATOMS, E = N_EDGES, B = N_MOLS, H = HDIM;

    // -------- inputs (setup_inputs dict order, params recursive insertion order)
    const float* x_in  = (const float*)d_in[0];
    const float* eatt  = (const float*)d_in[1];
    const int*   eidx  = (const int*)d_in[2];
    const int*   batch = (const int*)d_in[3];
    const int* src = eidx;          // edge_index[0]
    const int* dst = eidx + E;      // edge_index[1]

    int pi = 4;
    const float* atom_W = (const float*)d_in[pi++];
    const float* atom_b = (const float*)d_in[pi++];
    const float* bond_W = (const float*)d_in[pi++];
    const float* bond_b = (const float*)d_in[pi++];
    const float* lin1_W = (const float*)d_in[pi++];
    const float* lin1_b = (const float*)d_in[pi++];
    const float* gate_lin1_W = (const float*)d_in[pi++];
    const float* gate_lin2_W = (const float*)d_in[pi++];
    const float* gate_att_l  = (const float*)d_in[pi++];
    const float* gate_att_r  = (const float*)d_in[pi++];
    const float* gate_bias   = (const float*)d_in[pi++];
    const float* g0_Wih = (const float*)d_in[pi++];
    const float* g0_Whh = (const float*)d_in[pi++];
    const float* g0_bih = (const float*)d_in[pi++];
    const float* g0_bhh = (const float*)d_in[pi++];
    const float *cv_W[2], *cv_as[2], *cv_ad[2], *cv_b[2];
    for (int l = 0; l < 2; ++l) {
        cv_W[l]  = (const float*)d_in[pi++];
        cv_as[l] = (const float*)d_in[pi++];
        cv_ad[l] = (const float*)d_in[pi++];
        cv_b[l]  = (const float*)d_in[pi++];
    }
    const float *gr_Wih[2], *gr_Whh[2], *gr_bih[2], *gr_bhh[2];
    for (int l = 0; l < 2; ++l) {
        gr_Wih[l] = (const float*)d_in[pi++];
        gr_Whh[l] = (const float*)d_in[pi++];
        gr_bih[l] = (const float*)d_in[pi++];
        gr_bhh[l] = (const float*)d_in[pi++];
    }
    const float* mc_W  = (const float*)d_in[pi++];
    const float* mc_as = (const float*)d_in[pi++];
    const float* mc_ad = (const float*)d_in[pi++];
    const float* mc_b  = (const float*)d_in[pi++];
    const float* mg_Wih = (const float*)d_in[pi++];
    const float* mg_Whh = (const float*)d_in[pi++];
    const float* mg_bih = (const float*)d_in[pi++];
    const float* mg_bhh = (const float*)d_in[pi++];
    const float* lin2_W = (const float*)d_in[pi++];
    const float* lin2_b = (const float*)d_in[pi++];
    const float* proj_W = (const float*)d_in[pi++];
    const float* proj_b = (const float*)d_in[pi++];
    const float* ln_g   = (const float*)d_in[pi++];
    const float* ln_b   = (const float*)d_in[pi++];

    // -------- workspace layout (floats)
    float* Wsp = (float*)d_ws;
    size_t off = 0;
    auto alloc = [&](size_t n) { float* p = Wsp + off; off += n; return p; };
    float* bufX  = alloc((size_t)N * H);
    float* bufX2 = alloc((size_t)N * H);
    float* bufH  = alloc((size_t)N * H);
    float* bufXS = alloc((size_t)N * H);
    float* bufEA = alloc((size_t)E * EDGE_DIM);
    float* bufGI = alloc((size_t)N * G3);   // aliased as GATEConv g (E*256 < N*768)
    float* bufGH = alloc((size_t)N * G3);   // aliased as encoder pad staging (below)
    float* bufG  = bufGI;                   // alias: g dead before gi is written
    float* scoreE = alloc((size_t)E);       // reused for node scores (N<=E)
    float* wE     = alloc((size_t)E);
    unsigned* mEnc = (unsigned*)alloc((size_t)N);
    float* sSum = alloc((size_t)N);
    float* dR   = alloc((size_t)N);         // node dots (att_r / att_src / a_src)
    float* aDst = alloc((size_t)N);
    float* outA = alloc((size_t)B * H);
    float* outT = alloc((size_t)B * H);
    float* odB  = alloc((size_t)B * H);
    float* hB   = alloc((size_t)B * H);
    float* giB  = alloc((size_t)B * G3);
    float* ghB  = alloc((size_t)B * G3);
    // encoder pad staging aliases bufGH (dead until first GRU):
    // N*64 + 256*64 + E*32 + 64*32 = 12.84M floats << N*768 = 76.8M
    float* pX  = bufGH;                     // [N, 64]
    float* pAW = pX  + (size_t)N * ATOM_PAD;        // [256, 64]
    float* pEA = pAW + (size_t)HDIM * ATOM_PAD;     // [E, 32]
    float* pBW = pEA + (size_t)E * BOND_PAD;        // [64, 32]
    (void)ws_size; (void)in_sizes; (void)n_in; (void)out_size;

    dim3 bl256(256);
    auto waves8 = [&](long rows) { return dim3((unsigned)cdivl(rows, 8)); };
    auto thr    = [&](long n)    { return dim3((unsigned)cdivl(n, 256)); };

    // ===== encoders: zero-pad ragged K to 32-multiples, then fast WMMA =====
    padrows(x_in,   ATOM_DIM, pX,  ATOM_PAD, N, stream);
    padrows(atom_W, ATOM_DIM, pAW, ATOM_PAD, HDIM, stream);
    launch_fast(0, pX, ATOM_PAD, nullptr, nullptr, 0, 0,
                pAW, atom_b, bufX2, N, H, ATOM_PAD, stream);
    padrows(eatt,   BOND_DIM, pEA, BOND_PAD, E, stream);
    padrows(bond_W, BOND_DIM, pBW, BOND_PAD, EDGE_DIM, stream);
    launch_fast(0, pEA, BOND_PAD, nullptr, nullptr, 0, 0,
                pBW, bond_b, bufEA, E, EDGE_DIM, BOND_PAD, stream);
    // lin1 + leaky_relu
    launch_fast(1, bufX2, H, nullptr, nullptr, 0, 0,
                lin1_W, lin1_b, bufX, N, H, H, stream);

    // ===== GATEConv =====
    // g = lrelu( concat(x[src], ea) @ gate_lin1_W^T )  (gather + split-K GEMM)
    launch_fast(1, bufX, H, src, bufEA, EDGE_DIM, H,
                gate_lin1_W, nullptr, bufG, E, H, H + EDGE_DIM, stream);
    rowdot_kernel<<<waves8(N), bl256, 0, stream>>>(bufX, gate_att_r, dR, N, H);
    gate_score_kernel<<<waves8(E), bl256, 0, stream>>>(bufG, gate_att_l, dR, dst, scoreE, E, H);
    // segment softmax over dst (N segments)
    fillu(mEnc, ENC_NEGINF, N, stream);
    segmax_kernel<<<thr(E), bl256, 0, stream>>>(scoreE, dst, mEnc, E);
    fillf(sSum, 0.f, N, stream);
    segexp_kernel<<<thr(E), bl256, 0, stream>>>(scoreE, dst, mEnc, wE, sSum, E);
    // per-node xs2 = x @ gate_lin2^T, then weighted scatter to dst
    launch_fast(0, bufX, H, nullptr, nullptr, 0, 0,
                gate_lin2_W, nullptr, bufXS, N, H, H, stream);
    fillf(bufH, 0.f, (long)N * H, stream);
    scatter_rows_kernel<<<waves8(E), bl256, 0, stream>>>(bufXS, src, dst, wE, sSum, bufH, E, H);
    bias_act_kernel<0><<<dim3(N), bl256, 0, stream>>>(bufH, gate_bias);
    // GRU0: gi = h@Wih+bih ; gh = x@Whh+bhh ; combine (hprev = x)
    launch_fast(0, bufH, H, nullptr, nullptr, 0, 0, g0_Wih, g0_bih, bufGI, N, G3, H, stream);
    launch_fast(0, bufX, H, nullptr, nullptr, 0, 0, g0_Whh, g0_bhh, bufGH, N, G3, H, stream);
    gru_combine_kernel<<<dim3(N), bl256, 0, stream>>>(bufGI, bufGH, bufX, bufX2, H);

    // ===== 2x (GATConv + GRU) =====
    float* cur = bufX2;
    float* nxt = bufX;
    for (int l = 0; l < 2; ++l) {
        launch_fast(0, cur, H, nullptr, nullptr, 0, 0, cv_W[l], nullptr, bufXS, N, H, H, stream);
        rowdot_kernel<<<waves8(N), bl256, 0, stream>>>(bufXS, cv_as[l], dR, N, H);
        rowdot_kernel<<<waves8(N), bl256, 0, stream>>>(bufXS, cv_ad[l], aDst, N, H);
        score_kernel<<<thr(E), bl256, 0, stream>>>(dR, src, aDst, dst, scoreE, E);
        fillu(mEnc, ENC_NEGINF, N, stream);
        segmax_kernel<<<thr(E), bl256, 0, stream>>>(scoreE, dst, mEnc, E);
        fillf(sSum, 0.f, N, stream);
        segexp_kernel<<<thr(E), bl256, 0, stream>>>(scoreE, dst, mEnc, wE, sSum, E);
        fillf(bufH, 0.f, (long)N * H, stream);
        scatter_rows_kernel<<<waves8(E), bl256, 0, stream>>>(bufXS, src, dst, wE, sSum, bufH, E, H);
        bias_act_kernel<1><<<dim3(N), bl256, 0, stream>>>(bufH, cv_b[l]);
        launch_fast(0, bufH, H, nullptr, nullptr, 0, 0, gr_Wih[l], gr_bih[l], bufGI, N, G3, H, stream);
        launch_fast(0, cur,  H, nullptr, nullptr, 0, 0, gr_Whh[l], gr_bhh[l], bufGH, N, G3, H, stream);
        gru_combine_kernel<<<dim3(N), bl256, 0, stream>>>(bufGI, bufGH, cur, nxt, H);
        float* t = cur; cur = nxt; nxt = t;
    }

    // ===== attentive readout =====
    fillf(outA, 0.f, (long)B * H, stream);
    scatter_rows_kernel<<<waves8(N), bl256, 0, stream>>>(cur, nullptr, batch, nullptr, nullptr, outA, N, H);
    relu_kernel<<<thr((long)B * H), bl256, 0, stream>>>(outA, (long)B * H);
    launch_fast(0, cur, H, nullptr, nullptr, 0, 0, mc_W, nullptr, bufXS, N, H, H, stream);
    rowdot_kernel<<<waves8(N), bl256, 0, stream>>>(bufXS, mc_as, dR, N, H);   // a_src per node

    float* po = outA;
    float* pn = outT;
    for (int t = 0; t < 2; ++t) {
        launch_fast(0, po, H, nullptr, nullptr, 0, 0, mc_W, nullptr, odB, B, H, H, stream);
        rowdot_kernel<<<waves8(B), bl256, 0, stream>>>(odB, mc_ad, aDst, B, H);
        score_kernel<<<thr(N), bl256, 0, stream>>>(dR, nullptr, aDst, batch, scoreE, N);
        fillu(mEnc, ENC_NEGINF, B, stream);
        segmax_kernel<<<thr(N), bl256, 0, stream>>>(scoreE, batch, mEnc, N);
        fillf(sSum, 0.f, B, stream);
        segexp_kernel<<<thr(N), bl256, 0, stream>>>(scoreE, batch, mEnc, wE, sSum, N);
        fillf(hB, 0.f, (long)B * H, stream);
        scatter_rows_kernel<<<waves8(N), bl256, 0, stream>>>(bufXS, nullptr, batch, wE, sSum, hB, N, H);
        bias_act_kernel<0><<<dim3(B), bl256, 0, stream>>>(hB, mc_b);
        launch_fast(0, hB, H, nullptr, nullptr, 0, 0, mg_Wih, mg_bih, giB, B, G3, H, stream);
        launch_fast(0, po, H, nullptr, nullptr, 0, 0, mg_Whh, mg_bhh, ghB, B, G3, H, stream);
        gru_combine_kernel<<<dim3(B), bl256, 0, stream>>>(giB, ghB, po, pn, H);
        float* tp = po; po = pn; pn = tp;
    }

    // lin2 -> proj -> LayerNorm (in-place on d_out)
    launch_fast(0, po, H, nullptr, nullptr, 0, 0, lin2_W, lin2_b, odB, B, H, H, stream);
    float* emb = (float*)d_out;
    launch_fast(0, odB, H, nullptr, nullptr, 0, 0, proj_W, proj_b, emb, B, NODE_DIM, H, stream);
    layernorm_kernel<<<waves8(B), bl256, 0, stream>>>(emb, ln_g, ln_b, B);
}
